// DotAttention_10917806866534
// MI455X (gfx1250) — compile-verified, matrix operand order checked by reading, not run on hardware
//
#include <hip/hip_runtime.h>
#include <hip/hip_bf16.h>
#include <stdint.h>

// Problem constants (fixed by the reference)
#define NB 16
#define LQ 1024
#define LK 2048
#define DD 4096

// LDS row stride in u16 units: 64 payload + 4 pad (breaks bank conflicts,
// keeps rows 4B-aligned so paired-bf16 dword loads work)
#define LSTRIDE 68

typedef __attribute__((ext_vector_type(16))) __bf16 v16bf;
typedef __attribute__((ext_vector_type(8)))  float  v8f;
typedef __attribute__((ext_vector_type(4)))  unsigned int u32x4;
typedef __attribute__((ext_vector_type(8)))  int          i32x8;
typedef __attribute__((ext_vector_type(4)))  int          i32x4;

__device__ __forceinline__ unsigned short f2bf_bits(float f) {
  union { float f; unsigned u; } v; v.f = f;
  unsigned r = v.u + 0x7FFFu + ((v.u >> 16) & 1u);   // round-to-nearest-even
  return (unsigned short)(r >> 16);
}
__device__ __forceinline__ float bf2f(unsigned short s) {
  union { unsigned u; float f; } v; v.u = ((unsigned)s) << 16;
  return v.f;
}
__device__ __forceinline__ unsigned pack2(float lo, float hi) {
  return (unsigned)f2bf_bits(lo) | ((unsigned)f2bf_bits(hi) << 16);
}

// Assemble a 16x32 bf16 WMMA fragment (A or B) from LDS.
// CDNA5 ISA 7.12.2 (16-bit A 16x32, wave32):
//   lanes 0-15  (g=0): row = lane,    K in {0..7} U {16..23}
//   lanes 16-31 (g=1): row = lane-16, K in {8..15} U {24..31}
// VGPR v holds K pair: k = (v>=4?16:0) + g*8 + (v&3)*2
__device__ __forceinline__ v16bf frag_load(const unsigned short* lds,
                                           int row, int kbase, int g) {
  union { v16bf v; unsigned u[8]; } r;
  const unsigned short* base = lds + row * LSTRIDE;
#pragma unroll
  for (int vv = 0; vv < 8; ++vv) {
    int k = kbase + ((vv & 4) << 2) + g * 8 + (vv & 3) * 2;
    r.u[vv] = *(const unsigned*)(base + k);
  }
  return r.v;
}

#define WMMA_BF16(a, b, c) \
  __builtin_amdgcn_wmma_f32_16x16x32_bf16(false, (a), false, (b), (short)0, (c), false, false)

#if defined(__has_builtin)
#if __has_builtin(__builtin_amdgcn_tensor_load_to_lds) && __has_builtin(__builtin_amdgcn_s_wait_tensorcnt)
#define HAVE_TDM 1
#endif
#endif

#ifdef HAVE_TDM
// TDM: load a 2D tile (tile_w dwords x tile_h rows) of a row-major dword
// tensor into LDS, inserting 2 pad dwords after every 32 dwords so the LDS
// image matches the LSTRIDE=68-u16 padded layout used by frag_load.
// This toolchain exposes the 6-arg builtin:
//   (uint32x4 g0, int32x8 g1, int32x4 g2, int32x4 g3, int32x8, i32 cpol)
__device__ __forceinline__ void tdm_load_tile_pad(unsigned lds_off, const void* gsrc,
                                                  unsigned row_stride_u32,
                                                  unsigned tile_w_u32, unsigned tile_h,
                                                  unsigned tensor_w_u32, unsigned tensor_h) {
  unsigned long long ga = (unsigned long long)(uintptr_t)gsrc;
  u32x4 g0;
  g0[0] = 1u;                                                  // count=1, user D#
  g0[1] = lds_off;                                             // lds_addr (bytes)
  g0[2] = (unsigned)ga;                                        // global_addr[31:0]
  g0[3] = (unsigned)((ga >> 32) & 0x01FFFFFFu) | (2u << 30);   // addr[56:32] | type=2
  i32x8 g1;
  // data_size=2 (4B) | pad_enable | pad_interval=4 (32 dw) | pad_amount=1 (2 dw)
  g1[0] = (int)((2u << 16) | (1u << 20) | (4u << 22) | (1u << 25));
  g1[1] = (int)((tensor_w_u32 & 0xFFFFu) << 16);               // tensor_dim0[15:0]
  g1[2] = (int)((tensor_w_u32 >> 16) | ((tensor_h & 0xFFFFu) << 16));
  g1[3] = (int)((tensor_h >> 16) | (tile_w_u32 << 16));        // | tile_dim0
  g1[4] = (int)(tile_h & 0xFFFFu);                             // tile_dim1 (dim2=0)
  g1[5] = (int)row_stride_u32;                                 // tensor_dim0_stride lo
  g1[6] = 0;
  g1[7] = 0;
  i32x4 z4; z4[0] = 0; z4[1] = 0; z4[2] = 0; z4[3] = 0;
  i32x8 z8;
#pragma unroll
  for (int i = 0; i < 8; ++i) z8[i] = 0;
  __builtin_amdgcn_tensor_load_to_lds(g0, g1, z4, z4, z8, 0);
}
#endif

// ---------------------------------------------------------------------------
// Kernel 1: S[b,q,k] = sum_d Q[b,q,d] * K[b,k,d]   (bf16 WMMA, S -> ws bf16)
// Block tile 64(q) x 256(kk), BK=64 over D, double-buffered LDS.
// 8 waves: wave w -> rows (w&1)*32, cols (w>>1)*64; 2x4 grid of 16x16 accs.
// ---------------------------------------------------------------------------
__global__ __launch_bounds__(256)
void qk_gemm_kernel(const float* __restrict__ Q, const float* __restrict__ Kt,
                    unsigned short* __restrict__ S) {
  __shared__ unsigned short ldsQ[2][64  * LSTRIDE];   // [q][d]  bf16
  __shared__ unsigned short ldsK[2][256 * LSTRIDE];   // [kk][d] bf16

  const int b      = blockIdx.z;
  const int qbase  = blockIdx.y * 64;
  const int kkbase = blockIdx.x * 256;
  const int tid  = threadIdx.x;
  const int lane = tid & 31;
  const int wave = tid >> 5;
  const int g  = lane >> 4;
  const int nl = lane & 15;
  const int wm = wave & 1;
  const int wn = wave >> 1;

  const float* Qb = Q  + (size_t)b * LQ * DD + (size_t)qbase * DD;
  const float* Kb = Kt + (size_t)b * LK * DD + (size_t)kkbase * DD;

  v8f zero = {};
  v8f acc[2][4];
#pragma unroll
  for (int i = 0; i < 2; ++i)
#pragma unroll
    for (int j = 0; j < 4; ++j) acc[i][j] = zero;

  auto stage = [&](int d0, int buf) {
    // Q tile: 64x64 fp32 -> bf16 (1024 float4 slots, 4/thread)
#pragma unroll
    for (int it = 0; it < 4; ++it) {
      int linear = tid + it * 256;
      int row = linear >> 4;
      int dc  = (linear & 15) << 2;
      const float4 v = *(const float4*)(Qb + (size_t)row * DD + d0 + dc);
      unsigned* dst = (unsigned*)(&ldsQ[buf][0] + row * LSTRIDE + dc);
      dst[0] = pack2(v.x, v.y);
      dst[1] = pack2(v.z, v.w);
    }
    // K tile: 256x64 fp32 -> bf16 (4096 float4 slots, 16/thread)
#pragma unroll
    for (int it = 0; it < 16; ++it) {
      int linear = tid + it * 256;
      int row = linear >> 4;
      int dc  = (linear & 15) << 2;
      const float4 v = *(const float4*)(Kb + (size_t)row * DD + d0 + dc);
      unsigned* dst = (unsigned*)(&ldsK[buf][0] + row * LSTRIDE + dc);
      dst[0] = pack2(v.x, v.y);
      dst[1] = pack2(v.z, v.w);
    }
  };

  stage(0, 0);
  __syncthreads();

  const int NS = DD / 64;
#pragma clang loop unroll_count(2)
  for (int s = 0; s < NS; ++s) {
    const int cur = s & 1;
    if (s + 1 < NS) {
      __builtin_prefetch(Kb + (size_t)tid * DD + (s + 1) * 64, 0, 0);
      stage((s + 1) * 64, 1 - cur);
    }
    const unsigned short* Qc = &ldsQ[cur][0];
    const unsigned short* Kc = &ldsK[cur][0];
#pragma unroll
    for (int ks = 0; ks < 64; ks += 32) {
      v16bf a0 = frag_load(Qc, wm * 32 +      nl, ks, g);
      v16bf a1 = frag_load(Qc, wm * 32 + 16 + nl, ks, g);
      v16bf b0 = frag_load(Kc, wn * 64 +      nl, ks, g);
      v16bf b1 = frag_load(Kc, wn * 64 + 16 + nl, ks, g);
      v16bf b2 = frag_load(Kc, wn * 64 + 32 + nl, ks, g);
      v16bf b3 = frag_load(Kc, wn * 64 + 48 + nl, ks, g);
      acc[0][0] = WMMA_BF16(a0, b0, acc[0][0]);
      acc[0][1] = WMMA_BF16(a0, b1, acc[0][1]);
      acc[0][2] = WMMA_BF16(a0, b2, acc[0][2]);
      acc[0][3] = WMMA_BF16(a0, b3, acc[0][3]);
      acc[1][0] = WMMA_BF16(a1, b0, acc[1][0]);
      acc[1][1] = WMMA_BF16(a1, b1, acc[1][1]);
      acc[1][2] = WMMA_BF16(a1, b2, acc[1][2]);
      acc[1][3] = WMMA_BF16(a1, b3, acc[1][3]);
    }
    __syncthreads();
  }

  // Store S tile as bf16. C/D layout: VGPR r -> row (r + 8*g), col = nl.
  unsigned short* Sb = S + (size_t)b * LQ * LK;
#pragma unroll
  for (int i = 0; i < 2; ++i)
#pragma unroll
    for (int j = 0; j < 4; ++j) {
      int q0 = qbase + wm * 32 + i * 16 + 8 * g;
      int k0 = kkbase + wn * 64 + j * 16 + nl;
#pragma unroll
      for (int r = 0; r < 8; ++r)
        Sb[(size_t)(q0 + r) * LK + k0] = f2bf_bits(acc[i][j][r]);
    }
}

// ---------------------------------------------------------------------------
// Kernel 2: per-row softmax over Lk=2048 bf16 values, normalized P written
// back in place. One block (256 threads, 8 vals/thread) per (b,q) row.
// S is 64 MB bf16 -> L2-resident on the 192 MB global L2.
// ---------------------------------------------------------------------------
__global__ __launch_bounds__(256)
void softmax_row_kernel(unsigned short* __restrict__ S) {
  __shared__ float red[256];
  const int tid = threadIdx.x;
  unsigned short* p = S + (size_t)blockIdx.x * LK;

  float vals[8];
  const unsigned* p32 = (const unsigned*)p;
  unsigned u[4];
#pragma unroll
  for (int i = 0; i < 4; ++i) u[i] = p32[tid * 4 + i];
#pragma unroll
  for (int i = 0; i < 4; ++i) {
    vals[2 * i]     = bf2f((unsigned short)(u[i] & 0xFFFFu));
    vals[2 * i + 1] = bf2f((unsigned short)(u[i] >> 16));
  }

  float m = vals[0];
#pragma unroll
  for (int i = 1; i < 8; ++i) m = fmaxf(m, vals[i]);
  red[tid] = m;
  __syncthreads();
  for (int s = 128; s > 0; s >>= 1) {
    if (tid < s) red[tid] = fmaxf(red[tid], red[tid + s]);
    __syncthreads();
  }
  m = red[0];
  __syncthreads();

  float sum = 0.f;
#pragma unroll
  for (int i = 0; i < 8; ++i) { vals[i] = __expf(vals[i] - m); sum += vals[i]; }
  red[tid] = sum;
  __syncthreads();
  for (int s = 128; s > 0; s >>= 1) {
    if (tid < s) red[tid] += red[tid + s];
    __syncthreads();
  }
  const float inv = 1.0f / red[0];

  unsigned* q32 = (unsigned*)p;
#pragma unroll
  for (int i = 0; i < 4; ++i)
    q32[tid * 4 + i] = pack2(vals[2 * i] * inv, vals[2 * i + 1] * inv);
}

// ---------------------------------------------------------------------------
// Kernel 3: O[b,q,d] = sum_kk P[b,q,kk] * K[b,kk,d]   (bf16 WMMA, O fp32)
// Block tile 64(q) x 256(d), BK=64 over Lk, double-buffered LDS.
// P tile (already bf16) is DMA'd by the Tensor Data Mover with LDS padding
// matching LSTRIDE; K tile is transposed into LDS ([d][kk]) by the VALU path.
// ---------------------------------------------------------------------------
__global__ __launch_bounds__(256)
void pk_gemm_kernel(const unsigned short* __restrict__ P,
                    const float* __restrict__ Kt, float* __restrict__ O) {
  __shared__ unsigned short ldsP[2][64  * LSTRIDE];   // [q][kk] bf16
  __shared__ unsigned short ldsK[2][256 * LSTRIDE];   // [d][kk] bf16 (transposed)

  const int b     = blockIdx.z;
  const int qbase = blockIdx.y * 64;
  const int dbase = blockIdx.x * 256;
  const int tid  = threadIdx.x;
  const int lane = tid & 31;
  const int wave = tid >> 5;
  const int g  = lane >> 4;
  const int nl = lane & 15;
  const int wm = wave & 1;
  const int wn = wave >> 1;

  const unsigned short* Pb = P + (size_t)b * LQ * LK + (size_t)qbase * LK;
  const float* Kb = Kt + (size_t)b * LK * DD;

  v8f zero = {};
  v8f acc[2][4];
#pragma unroll
  for (int i = 0; i < 2; ++i)
#pragma unroll
    for (int j = 0; j < 4; ++j) acc[i][j] = zero;

  auto stageP = [&](int k0, int buf) {
#ifdef HAVE_TDM
    if (wave == 0)
      tdm_load_tile_pad((unsigned)(size_t)(const void*)&ldsP[buf][0],
                        (const void*)(Pb + k0),
                        LK / 2,        // row stride in dwords
                        32, 64,        // tile: 32 dwords x 64 rows
                        LK / 2, NB * LQ);
#else
#pragma unroll
    for (int it = 0; it < 8; ++it) {
      int linear = tid + it * 256;             // 0..2047
      int row = linear >> 5;                   // q: 0..63
      int c   = linear & 31;                   // dword col (kk pair)
      unsigned v = *(const unsigned*)(Pb + (size_t)row * LK + k0 + c * 2);
      *(unsigned*)(&ldsP[buf][0] + row * LSTRIDE + c * 2) = v;
    }
#endif
  };
  auto stageK = [&](int k0, int buf) {
    // 64(kk) x 256(d) fp32 -> bf16 transposed [d][kk] (4096 float4, 16/thread)
#pragma unroll
    for (int it = 0; it < 16; ++it) {
      int linear = tid + it * 256;             // 0..4095
      int row = linear >> 6;                   // kk: 0..63
      int c4  = (linear & 63) << 2;            // d: 0..252 step 4
      const float4 v = *(const float4*)(Kb + (size_t)(k0 + row) * DD + dbase + c4);
      unsigned short* dst = &ldsK[buf][0];
      dst[(c4 + 0) * LSTRIDE + row] = f2bf_bits(v.x);
      dst[(c4 + 1) * LSTRIDE + row] = f2bf_bits(v.y);
      dst[(c4 + 2) * LSTRIDE + row] = f2bf_bits(v.z);
      dst[(c4 + 3) * LSTRIDE + row] = f2bf_bits(v.w);
    }
  };

  stageP(0, 0);
  stageK(0, 0);
#ifdef HAVE_TDM
  if (wave == 0) __builtin_amdgcn_s_wait_tensorcnt(0);
#endif
  __syncthreads();

  const int NS = LK / 64;
#pragma clang loop unroll_count(2)
  for (int s = 0; s < NS; ++s) {
    const int cur = s & 1;
    if (s + 1 < NS) {
      stageP((s + 1) * 64, 1 - cur);
      stageK((s + 1) * 64, 1 - cur);
    }
    const unsigned short* Pc = &ldsP[cur][0];
    const unsigned short* Kc = &ldsK[cur][0];
#pragma unroll
    for (int ks = 0; ks < 64; ks += 32) {
      v16bf a0 = frag_load(Pc, wm * 32 +      nl, ks, g);
      v16bf a1 = frag_load(Pc, wm * 32 + 16 + nl, ks, g);
      v16bf b0 = frag_load(Kc, wn * 64 +      nl, ks, g);
      v16bf b1 = frag_load(Kc, wn * 64 + 16 + nl, ks, g);
      v16bf b2 = frag_load(Kc, wn * 64 + 32 + nl, ks, g);
      v16bf b3 = frag_load(Kc, wn * 64 + 48 + nl, ks, g);
      acc[0][0] = WMMA_BF16(a0, b0, acc[0][0]);
      acc[0][1] = WMMA_BF16(a0, b1, acc[0][1]);
      acc[0][2] = WMMA_BF16(a0, b2, acc[0][2]);
      acc[0][3] = WMMA_BF16(a0, b3, acc[0][3]);
      acc[1][0] = WMMA_BF16(a1, b0, acc[1][0]);
      acc[1][1] = WMMA_BF16(a1, b1, acc[1][1]);
      acc[1][2] = WMMA_BF16(a1, b2, acc[1][2]);
      acc[1][3] = WMMA_BF16(a1, b3, acc[1][3]);
    }
#ifdef HAVE_TDM
    if (s + 1 < NS && wave == 0) __builtin_amdgcn_s_wait_tensorcnt(0);
#endif
    __syncthreads();
  }

  float* Ob = O + (size_t)b * LQ * DD;
#pragma unroll
  for (int i = 0; i < 2; ++i)
#pragma unroll
    for (int j = 0; j < 4; ++j) {
      int q0 = qbase + wm * 32 + i * 16 + 8 * g;
      int d0 = dbase + wn * 64 + j * 16 + nl;
#pragma unroll
      for (int r = 0; r < 8; ++r)
        Ob[(size_t)(q0 + r) * DD + d0] = acc[i][j][r];
    }
}

extern "C" void kernel_launch(void* const* d_in, const int* in_sizes, int n_in,
                              void* d_out, int out_size, void* d_ws, size_t ws_size,
                              hipStream_t stream) {
  const float* Q = (const float*)d_in[0];   // [16,1024,4096] fp32
  const float* K = (const float*)d_in[1];   // [16,2048,4096] fp32
  float* out = (float*)d_out;               // [16,1024,4096] fp32
  // Workspace: S/P as bf16, 16*1024*2048*2 = 64 MB (L2-resident on 192MB L2)
  unsigned short* S = (unsigned short*)d_ws;

  dim3 blk(256, 1, 1);
  qk_gemm_kernel<<<dim3(LK / 256, LQ / 64, NB), blk, 0, stream>>>(Q, K, S);
  softmax_row_kernel<<<dim3(NB * LQ, 1, 1), blk, 0, stream>>>(S);
  pk_gemm_kernel<<<dim3(DD / 256, LQ / 64, NB), blk, 0, stream>>>(S, K, out);
}
